// LagSpectrumAttention_29506425323859
// MI455X (gfx1250) — matrix operand order repaired and checked by reading, not compiled, not run on hardware
//
#include <hip/hip_runtime.h>
#include <hip/hip_bf16.h>

typedef __attribute__((ext_vector_type(16))) __bf16 v16bf;
typedef __attribute__((ext_vector_type(8)))  float  v8f;

#define D_MODEL 512
#define N_HEADS 8
#define D_K     64
#define B_SZ    4
#define T_SZ    128
#define N_ENT   64
#define MIN_LAG 2
#define MAX_LAG 12
#define N_LAGS  11   // MAX_LAG - MIN_LAG + 1
#define MAXLEN  126  // T - MIN_LAG

// float -> bf16 (RNE via hardware cvt, lowers to v_cvt_pk_bf16_f32 pairs)
__device__ __forceinline__ __bf16 f2bf(float f) { return (__bf16)f; }

// A-matrix (16x32 bf16) K index for element e within the v16bf, per lane half.
// Contiguous order K = 16*half + e, matching the contiguous B fragment reads.
__device__ __forceinline__ int a_kpat(int e, int half) {
    return 16 * half + e;
}

__device__ __forceinline__ v16bf lds_frag(const __bf16* p) {
    return *(const v16bf*)p;  // 32B aligned LDS read -> 2x ds_load_b128
}

// ---------------------------------------------------------------------------
// Out[m, n] = sum_d X[m, d] * W[n, d]    (M x 512) @ (512 x 512)^T
// grid = (M/64, 512/64), block = 128 (4 waves). The 32x64 W tile is shared by
// all 4 waves -> stage it once per K-step in LDS as bf16 (4x less global
// traffic + 4x fewer cvts). A rows are wave-private -> load from global.
// ---------------------------------------------------------------------------
__global__ void gemm_xwt_bf16(const float* __restrict__ X,
                              const float* __restrict__ W,
                              float* __restrict__ Out) {
    __shared__ alignas(32) __bf16 WB[64][32];  // [n][k] bf16, 4 KB

    const int tid  = threadIdx.x;
    const int wave = tid >> 5;
    const int lane = tid & 31;
    const int col  = lane & 15;
    const int half = lane >> 4;
    const int m0 = blockIdx.x * 64 + wave * 16;
    const int n0 = blockIdx.y * 64;

    // cooperative-fill coordinates: thread t -> row n = t/2, k-half = (t&1)*16
    const int fn = tid >> 1;
    const int fk = (tid & 1) * 16;

    v8f acc[4] = {{}, {}, {}, {}};

    for (int k0 = 0; k0 < D_MODEL; k0 += 32) {
        __syncthreads();  // previous step's fragment reads done
        {
            const float* wsrc = W + (size_t)(n0 + fn) * D_MODEL + k0 + fk;
            #pragma unroll
            for (int i = 0; i < 16; ++i) WB[fn][fk + i] = f2bf(wsrc[i]);
        }
        __syncthreads();

        v16bf a;
        const float* xrow = X + (size_t)(m0 + col) * D_MODEL + k0;
        #pragma unroll
        for (int e = 0; e < 16; ++e) a[e] = f2bf(xrow[a_kpat(e, half)]);

        #pragma unroll
        for (int nt = 0; nt < 4; ++nt) {
            v16bf b = lds_frag(&WB[nt * 16 + col][16 * half]);
            acc[nt] = __builtin_amdgcn_wmma_f32_16x16x32_bf16(
                false, a, false, b, (short)0, acc[nt], false, false);
        }
    }
    #pragma unroll
    for (int nt = 0; nt < 4; ++nt)
        #pragma unroll
        for (int r = 0; r < 8; ++r)
            Out[(size_t)(m0 + r + 8 * half) * D_MODEL + n0 + nt * 16 + col] =
                acc[nt][r];
}

// ---------------------------------------------------------------------------
// Zero the t=0,1 slices of combined (never written by attention).
// ---------------------------------------------------------------------------
__global__ void zero_head_rows(float* __restrict__ comb) {
    const int per_b = 2 * N_ENT * D_MODEL;  // 65536
    int idx = blockIdx.x * blockDim.x + threadIdx.x;
    int b = idx / per_b;
    int r = idx - b * per_b;
    if (b < B_SZ)
        comb[(size_t)b * T_SZ * N_ENT * D_MODEL + r] = 0.0f;
}

// ---------------------------------------------------------------------------
// Lag attention. One 128-thread block per (b, tq, head), tq = 2..127.
// Wave w owns score/output rows i0 = 16w..16w+15. Per lag, the 64x64 K_l
// (+lag_emb) and V_l tiles are identical for all 4 waves -> stage once in
// LDS as bf16 (V transposed so both fragment reads are contiguous 32B).
// Scores + softmax staged in LDS; output accumulated fp32 across lags.
// ---------------------------------------------------------------------------
__global__ void lag_attention(const float* __restrict__ Q,
                              const float* __restrict__ K,
                              const float* __restrict__ V,
                              const float* __restrict__ lag_emb,
                              float* __restrict__ Comb) {
    __shared__ float S[N_ENT * N_ENT];                   // 16 KB scores/probs
    __shared__ alignas(32) __bf16 KB[N_ENT][D_K];        // 8 KB: [j][c]
    __shared__ alignas(32) __bf16 VBT[D_K][N_ENT];       // 8 KB: [c][j]

    const int tid  = threadIdx.x;
    const int wave = tid >> 5;
    const int lane = tid & 31;
    const int col  = lane & 15;
    const int half = lane >> 4;

    const int blk = blockIdx.x;
    const int h  = blk & 7;
    const int tq = 2 + ((blk >> 3) % MAXLEN);
    const int b  = blk / (MAXLEN * N_HEADS);

    const int i0 = wave * 16;
    const size_t qbase =
        ((size_t)(b * T_SZ + tq) * N_ENT) * D_MODEL + (size_t)h * D_K;

    // cooperative-fill coordinates: thread t -> row j = t/2, c-half = (t&1)*32
    const int fj = tid >> 1;
    const int fc = (tid & 1) * 32;

    // Q A-fragments (rows i0..i0+15, c split into two 32-chunks) — lag-
    // invariant, load once from global.
    v16bf aQ[2];
    #pragma unroll
    for (int kc = 0; kc < 2; ++kc) {
        const float* qrow = Q + qbase + (size_t)(i0 + col) * D_MODEL + kc * 32;
        #pragma unroll
        for (int e = 0; e < 16; ++e) aQ[kc][e] = f2bf(qrow[a_kpat(e, half)]);
    }

    v8f o[4] = {{}, {}, {}, {}};

    const int lag_max = (tq < MAX_LAG) ? tq : MAX_LAG;
    for (int lag = MIN_LAG; lag <= lag_max; ++lag) {
        const int tk = tq - lag;
        const int li = lag - MIN_LAG;
        const size_t kvbase =
            ((size_t)(b * T_SZ + tk) * N_ENT) * D_MODEL + (size_t)h * D_K;
        const float* le = lag_emb + (size_t)li * D_MODEL + h * D_K;

        __syncthreads();  // previous lag's P@V reads of VBT/S done
        {   // stage K_l + lag_emb as [j][c]; V_l transposed as [c][j]
            const float* ksrc = K + kvbase + (size_t)fj * D_MODEL + fc;
            const float* vsrc = V + kvbase + (size_t)fj * D_MODEL + fc;
            const float* lsrc = le + fc;
            #pragma unroll
            for (int c = 0; c < 32; ++c) {
                KB[fj][fc + c] = f2bf(ksrc[c] + lsrc[c]);
                VBT[fc + c][fj] = f2bf(vsrc[c]);
            }
        }
        __syncthreads();

        // S = Q @ (K + lag_emb)^T : B col j = lane%16, k-dim = c.
        v8f s[4] = {{}, {}, {}, {}};
        #pragma unroll
        for (int jt = 0; jt < 4; ++jt) {
            #pragma unroll
            for (int kc = 0; kc < 2; ++kc) {
                v16bf bK = lds_frag(&KB[jt * 16 + col][kc * 32 + 16 * half]);
                s[jt] = __builtin_amdgcn_wmma_f32_16x16x32_bf16(
                    false, aQ[kc], false, bK, (short)0, s[jt], false, false);
            }
        }
        #pragma unroll
        for (int jt = 0; jt < 4; ++jt)
            #pragma unroll
            for (int r = 0; r < 8; ++r)
                S[(i0 + r + 8 * half) * N_ENT + jt * 16 + col] =
                    s[jt][r] * 0.125f;  // / sqrt(D_K)
        __syncthreads();

        // Row softmax: threads 0..63 each own one row.
        if (tid < N_ENT) {
            float* row = S + tid * N_ENT;
            float m = row[0];
            for (int j = 1; j < N_ENT; ++j) m = fmaxf(m, row[j]);
            float sum = 0.0f;
            for (int j = 0; j < N_ENT; ++j) {
                float e = __expf(row[j] - m);
                row[j] = e;
                sum += e;
            }
            float inv = 1.0f / sum;
            for (int j = 0; j < N_ENT; ++j) row[j] *= inv;
        }
        __syncthreads();

        // O += P @ V : A = P (16x32 chunks over j), B = V^T staged as [c][j].
        #pragma unroll
        for (int jc = 0; jc < 2; ++jc) {
            v16bf aP;
            const float* prow = S + (i0 + col) * N_ENT + jc * 32;
            #pragma unroll
            for (int e = 0; e < 16; ++e) aP[e] = f2bf(prow[a_kpat(e, half)]);
            #pragma unroll
            for (int ct = 0; ct < 4; ++ct) {
                v16bf bV =
                    lds_frag(&VBT[ct * 16 + col][jc * 32 + 16 * half]);
                o[ct] = __builtin_amdgcn_wmma_f32_16x16x32_bf16(
                    false, aP, false, bV, (short)0, o[ct], false, false);
            }
        }
    }

    const float invL = 1.0f / (float)N_LAGS;  // stacked mean over 11 lags
    #pragma unroll
    for (int ct = 0; ct < 4; ++ct)
        #pragma unroll
        for (int r = 0; r < 8; ++r)
            Comb[qbase + (size_t)(i0 + r + 8 * half) * D_MODEL +
                 ct * 16 + col] = o[ct][r] * invL;
}

// ---------------------------------------------------------------------------
extern "C" void kernel_launch(void* const* d_in, const int* in_sizes, int n_in,
                              void* d_out, int out_size, void* d_ws,
                              size_t ws_size, hipStream_t stream) {
    (void)in_sizes; (void)n_in; (void)out_size; (void)ws_size;
    const float* x       = (const float*)d_in[0];
    const float* Wq      = (const float*)d_in[1];
    const float* Wk      = (const float*)d_in[2];
    const float* Wv      = (const float*)d_in[3];
    const float* Wo      = (const float*)d_in[4];
    const float* lag_emb = (const float*)d_in[5];
    float* out = (float*)d_out;

    const size_t MAT = (size_t)B_SZ * T_SZ * N_ENT * D_MODEL;  // 16M floats
    float* qb = (float*)d_ws;
    float* kb = qb + MAT;
    float* vb = kb + MAT;
    float* cb = vb + MAT;

    const dim3 ggrid(B_SZ * T_SZ * N_ENT / 64, D_MODEL / 64);  // (512, 8)
    const dim3 gblk(128);

    gemm_xwt_bf16<<<ggrid, gblk, 0, stream>>>(x, Wq, qb);
    gemm_xwt_bf16<<<ggrid, gblk, 0, stream>>>(x, Wk, kb);
    gemm_xwt_bf16<<<ggrid, gblk, 0, stream>>>(x, Wv, vb);

    zero_head_rows<<<(B_SZ * 2 * N_ENT * D_MODEL) / 256, 256, 0, stream>>>(cb);

    lag_attention<<<B_SZ * MAXLEN * N_HEADS, 128, 0, stream>>>(
        qb, kb, vb, lag_emb, cb);

    gemm_xwt_bf16<<<ggrid, gblk, 0, stream>>>(cb, Wo, out);
}
